// S2VT_30545807409873
// MI455X (gfx1250) — matrix operand order repaired and checked by reading, not compiled, not run on hardware
//
#include <hip/hip_runtime.h>
#include <hip/hip_bf16.h>
#include <stdint.h>

#define B_   128
#define T_   80
#define I_   4096
#define H_   256
#define H4_  1024
#define V_   6000
#define S_   40

typedef __attribute__((ext_vector_type(16))) __bf16 v16bf;
typedef __attribute__((ext_vector_type(8)))  __bf16 v8bf;
typedef __attribute__((ext_vector_type(8)))  float  v8f;

// ---------------- WMMA helpers (CDNA5 gfx1250, wave32) ----------------

__device__ __forceinline__ v8f wmma_bf16(v16bf a, v16bf b, v8f c) {
  // D(f32 16x16) = A(16x32 bf16) * B(32x16 bf16) + C
  return __builtin_amdgcn_wmma_f32_16x16x32_bf16(
      /*neg_a=*/false, a, /*neg_b=*/false, b,
      /*c_mod=*/(short)0, c, /*reuse_a=*/false, /*reuse_b=*/false);
}

// A-matrix 16x32 bf16 fragment from row-major [rows x ld] source.
// Lane L (L&15 = row). VGPR0..3: K = k+{0..7} (+8 if L>=16); VGPR4..7: K += 16.
__device__ __forceinline__ v16bf load_a_frag(const __bf16* __restrict__ base, int ld,
                                             int row_base, int k_base, int row_max) {
  int lane = threadIdx.x & 31;
  int r = row_base + (lane & 15);
  if (r >= row_max) r = row_max - 1;
  int k = k_base + ((lane >> 4) << 3);
  const __bf16* p = base + (size_t)r * (size_t)ld + k;
  v8bf lo = *(const v8bf*)(p);
  v8bf hi = *(const v8bf*)(p + 16);
  v16bf out;
#pragma unroll
  for (int i = 0; i < 8; ++i) { out[i] = lo[i]; out[i + 8] = hi[i]; }
  return out;
}

// B-matrix 32x16 bf16 fragment; source is row-major W [N x K] (we compute X*W^T,
// so B column n == W row n). Same K order as A: K = k+{0..7} (+8 if L>=16); elems 8..15: K += 16.
__device__ __forceinline__ v16bf load_b_frag(const __bf16* __restrict__ base, int ld,
                                             int col_base, int k_base, int col_max) {
  int lane = threadIdx.x & 31;
  int n = col_base + (lane & 15);
  if (n >= col_max) n = col_max - 1;
  int k = k_base + ((lane >> 4) << 3);
  const __bf16* p = base + (size_t)n * (size_t)ld + k;
  v8bf lo = *(const v8bf*)(p);
  v8bf hi = *(const v8bf*)(p + 16);
  v16bf out;
#pragma unroll
  for (int i = 0; i < 8; ++i) { out[i] = lo[i]; out[i + 8] = hi[i]; }
  return out;
}

// ---------------- Small WMMA GEMM (recurrent steps): Out = A @ W^T [+Csrc] [+bias]
// Block = 256 threads = 8 waves -> 128(M) x 64(N); wave tile 32x32 (2x2 frags).
// Chosen for grid spread (M=128,N=1024 -> 16 blocks) on the latency-bound chain.

__global__ void __launch_bounds__(256)
wmma_gemm_k(const __bf16* __restrict__ A, int lda,
            const __bf16* __restrict__ W, int ldw,
            const float* __restrict__ Csrc, long long ldc,
            const float* __restrict__ bias,
            float* __restrict__ Out, long long ldo,
            int M, int N, int K) {
  const int wave = threadIdx.x >> 5;      // 0..7
  const int wm = wave >> 1;               // 0..3 along M
  const int wn = wave & 1;                // 0..1 along N
  const int m0 = blockIdx.x * 128 + wm * 32;
  const int n0 = blockIdx.y * 64  + wn * 32;

  v8f acc[2][2] = {{{0.f},{0.f}},{{0.f},{0.f}}};

  for (int k = 0; k < K; k += 32) {
    v16bf a0 = load_a_frag(A, lda, m0,      k, M);
    v16bf a1 = load_a_frag(A, lda, m0 + 16, k, M);
    v16bf b0 = load_b_frag(W, ldw, n0,      k, N);
    v16bf b1 = load_b_frag(W, ldw, n0 + 16, k, N);
    acc[0][0] = wmma_bf16(a0, b0, acc[0][0]);
    acc[0][1] = wmma_bf16(a0, b1, acc[0][1]);
    acc[1][0] = wmma_bf16(a1, b0, acc[1][0]);
    acc[1][1] = wmma_bf16(a1, b1, acc[1][1]);
  }

  // C/D layout: lane&15 = N column; lane>=16 -> M += 8; VGPR r -> M row r.
  const int lane = threadIdx.x & 31;
  const int ncol = lane & 15;
  const int mhi  = (lane >> 4) * 8;
#pragma unroll
  for (int im = 0; im < 2; ++im) {
#pragma unroll
    for (int in = 0; in < 2; ++in) {
      int n = n0 + in * 16 + ncol;
      if (n >= N) continue;
      float badd = bias ? bias[n] : 0.f;
#pragma unroll
      for (int r = 0; r < 8; ++r) {
        int m = m0 + im * 16 + mhi + r;
        if (m >= M) continue;
        float v = acc[im][in][r] + badd;
        if (Csrc) v += Csrc[(long long)m * ldc + n];
        Out[(long long)m * ldo + n] = v;
      }
    }
  }
}

// ---------------- Big WMMA GEMM (throughput): Out = A @ W^T [+bias]
// Block = 256 threads = 8 waves (4Mx2N) -> 128(M) x 128(N); wave tile 32x64
// (2x4 frags, 8 WMMAs per 6 fragment loads -> 1.5 b128 loads/WMMA).

__global__ void __launch_bounds__(256)
wmma_gemm_big_k(const __bf16* __restrict__ A, int lda,
                const __bf16* __restrict__ W, int ldw,
                const float* __restrict__ bias,
                float* __restrict__ Out, long long ldo,
                int M, int N, int K) {
  const int wave = threadIdx.x >> 5;      // 0..7
  const int wm = wave >> 1;               // 0..3 along M
  const int wn = wave & 1;                // 0..1 along N
  const int m0 = blockIdx.x * 128 + wm * 32;
  const int n0 = blockIdx.y * 128 + wn * 64;

  v8f acc[2][4];
#pragma unroll
  for (int i = 0; i < 2; ++i)
#pragma unroll
    for (int j = 0; j < 4; ++j) acc[i][j] = (v8f){0.f};

  for (int k = 0; k < K; k += 32) {
    v16bf a0 = load_a_frag(A, lda, m0,      k, M);
    v16bf a1 = load_a_frag(A, lda, m0 + 16, k, M);
    v16bf b0 = load_b_frag(W, ldw, n0,      k, N);
    v16bf b1 = load_b_frag(W, ldw, n0 + 16, k, N);
    v16bf b2 = load_b_frag(W, ldw, n0 + 32, k, N);
    v16bf b3 = load_b_frag(W, ldw, n0 + 48, k, N);
    acc[0][0] = wmma_bf16(a0, b0, acc[0][0]);
    acc[0][1] = wmma_bf16(a0, b1, acc[0][1]);
    acc[0][2] = wmma_bf16(a0, b2, acc[0][2]);
    acc[0][3] = wmma_bf16(a0, b3, acc[0][3]);
    acc[1][0] = wmma_bf16(a1, b0, acc[1][0]);
    acc[1][1] = wmma_bf16(a1, b1, acc[1][1]);
    acc[1][2] = wmma_bf16(a1, b2, acc[1][2]);
    acc[1][3] = wmma_bf16(a1, b3, acc[1][3]);
  }

  const int lane = threadIdx.x & 31;
  const int ncol = lane & 15;
  const int mhi  = (lane >> 4) * 8;
#pragma unroll
  for (int im = 0; im < 2; ++im) {
#pragma unroll
    for (int in = 0; in < 4; ++in) {
      int n = n0 + in * 16 + ncol;
      if (n >= N) continue;
      float badd = bias ? bias[n] : 0.f;
#pragma unroll
      for (int r = 0; r < 8; ++r) {
        int m = m0 + im * 16 + mhi + r;
        if (m >= M) continue;
        Out[(long long)m * ldo + n] = acc[im][in][r] + badd;
      }
    }
  }
}

// ---------------- Elementwise LSTM cell (gate order i,f,g,o) ----------------

__device__ __forceinline__ float sigm(float x) { return 1.0f / (1.0f + __expf(-x)); }

__global__ void __launch_bounds__(256)
lstm_cell_k(const float* __restrict__ gates,  // B x 4H
            float* __restrict__ c,            // B x H (in/out)
            __bf16* __restrict__ hb1, long long ld1,   // bf16 h dest #1 (nullable)
            __bf16* __restrict__ hb2, long long ld2,   // bf16 dest #2 (nullable)
            const float* __restrict__ addsrc, // added to h for hb2 only (nullable)
            float* __restrict__ hf) {         // fp32 h dest (nullable)
  int idx = blockIdx.x * blockDim.x + threadIdx.x;
  if (idx >= B_ * H_) return;
  int b = idx >> 8;         // /H_
  int j = idx & (H_ - 1);
  const float* g = gates + (long long)b * H4_;
  float gi = sigm(g[j]);
  float gf = sigm(g[H_ + j]);
  float gg = tanhf(g[2 * H_ + j]);
  float go = sigm(g[3 * H_ + j]);
  float cn = gf * c[idx] + gi * gg;
  c[idx] = cn;
  float h = go * tanhf(cn);
  if (hb1) hb1[(long long)b * ld1 + j] = (__bf16)h;
  if (hb2) {
    float h2 = h + (addsrc ? addsrc[idx] : 0.f);
    hb2[(long long)b * ld2 + j] = (__bf16)h2;
  }
  if (hf) hf[idx] = h;
}

// ---------------- Small utility kernels ----------------

__global__ void __launch_bounds__(256)
cast4_bf16_k(const float* __restrict__ s, __bf16* __restrict__ d, long long n4) {
  long long i = (long long)blockIdx.x * blockDim.x + threadIdx.x;
  if (i >= n4) return;
  float4 v = ((const float4*)s)[i];
  __bf16* o = d + 4 * i;
  o[0] = (__bf16)v.x; o[1] = (__bf16)v.y; o[2] = (__bf16)v.z; o[3] = (__bf16)v.w;
}

__global__ void __launch_bounds__(256)
build_w2cat_k(const float* __restrict__ wih2, const float* __restrict__ whh2,
              __bf16* __restrict__ dst) {
  int i = blockIdx.x * blockDim.x + threadIdx.x;       // over H4_*2H_
  if (i >= H4_ * 2 * H_) return;
  int r = i / (2 * H_), j = i % (2 * H_);
  float v = (j < H_) ? wih2[r * H_ + j] : whh2[r * H_ + (j - H_)];
  dst[i] = (__bf16)v;
}

__global__ void __launch_bounds__(256)
vec_add_k(const float* __restrict__ a, const float* __restrict__ b,
          float* __restrict__ o, int n) {
  int i = blockIdx.x * blockDim.x + threadIdx.x;
  if (i < n) o[i] = a[i] + b[i];
}

__global__ void __launch_bounds__(256)
zero_f32_k(float* p, int n) {
  int i = blockIdx.x * blockDim.x + threadIdx.x;
  if (i < n) p[i] = 0.f;
}

__global__ void __launch_bounds__(256)
zero_bf16_k(__bf16* p, int n) {
  int i = blockIdx.x * blockDim.x + threadIdx.x;
  if (i < n) p[i] = (__bf16)0.f;
}

// ---------------- Host orchestration ----------------

static inline char* wsa(char*& p, size_t bytes) {
  char* r = p;
  p += (bytes + 255) & ~(size_t)255;
  return r;
}

static inline void launch_gemm(hipStream_t st,
                               const __bf16* A, int lda, const __bf16* W, int ldw,
                               const float* Csrc, long long ldc, const float* bias,
                               float* Out, long long ldo, int M, int N, int K) {
  dim3 grid((M + 127) / 128, (N + 63) / 64);
  wmma_gemm_k<<<grid, 256, 0, st>>>(A, lda, W, ldw, Csrc, ldc, bias, Out, ldo, M, N, K);
}

static inline void launch_gemm_big(hipStream_t st,
                                   const __bf16* A, int lda, const __bf16* W, int ldw,
                                   const float* bias,
                                   float* Out, long long ldo, int M, int N, int K) {
  dim3 grid((M + 127) / 128, (N + 127) / 128);
  wmma_gemm_big_k<<<grid, 256, 0, st>>>(A, lda, W, ldw, bias, Out, ldo, M, N, K);
}

extern "C" void kernel_launch(void* const* d_in, const int* in_sizes, int n_in,
                              void* d_out, int out_size, void* d_ws, size_t ws_size,
                              hipStream_t stream) {
  const float* x     = (const float*)d_in[0];   // (B,T,I)
  const float* Wih1  = (const float*)d_in[1];   // (4H,I)
  const float* Whh1  = (const float*)d_in[2];   // (4H,H)
  const float* bih1  = (const float*)d_in[3];
  const float* bhh1  = (const float*)d_in[4];
  const float* Wih2  = (const float*)d_in[5];   // (4H,H)
  const float* Whh2  = (const float*)d_in[6];   // (4H,H)
  const float* bih2  = (const float*)d_in[7];
  const float* bhh2  = (const float*)d_in[8];
  const float* Wout  = (const float*)d_in[9];   // (V,H)
  const float* bout  = (const float*)d_in[10];  // (V)
  float* out = (float*)d_out;                   // (B*S, V)

  char* p = (char*)d_ws;
  __bf16* x_bf    = (__bf16*)wsa(p, (size_t)B_ * T_ * I_ * 2);   // 84 MB
  __bf16* Wih1_bf = (__bf16*)wsa(p, (size_t)H4_ * I_ * 2);
  __bf16* Whh1_bf = (__bf16*)wsa(p, (size_t)H4_ * H_ * 2);
  __bf16* W2c_bf  = (__bf16*)wsa(p, (size_t)H4_ * 2 * H_ * 2);   // [Wih2 | Whh2]
  __bf16* Wout_bf = (__bf16*)wsa(p, (size_t)V_ * H_ * 2);
  float*  G1      = (float*) wsa(p, (size_t)B_ * T_ * H4_ * 4);  // 42 MB, row = b*T+t
  float*  gates1  = (float*) wsa(p, (size_t)B_ * H4_ * 4);
  float*  gates2  = (float*) wsa(p, (size_t)B_ * H4_ * 4);
  float*  c1      = (float*) wsa(p, (size_t)B_ * H_ * 4);
  float*  c2      = (float*) wsa(p, (size_t)B_ * H_ * 4);
  float*  y2      = (float*) wsa(p, (size_t)B_ * H_ * 4);
  __bf16* h1b     = (__bf16*)wsa(p, (size_t)B_ * H_ * 2);        // h1 (pure)
  __bf16* xcat    = (__bf16*)wsa(p, (size_t)B_ * 2 * H_ * 2);    // [lstm2 input | h2]
  __bf16* feats   = (__bf16*)wsa(p, (size_t)B_ * S_ * H_ * 2);   // row = b*S+s
  float*  b1s     = (float*) wsa(p, H4_ * 4);
  float*  b2s     = (float*) wsa(p, H4_ * 4);
  (void)ws_size; (void)in_sizes; (void)n_in; (void)out_size;

  // ---- init / casts ----
  {
    long long n4;
    n4 = (long long)B_ * T_ * I_ / 4;
    cast4_bf16_k<<<(unsigned)((n4 + 255) / 256), 256, 0, stream>>>(x, x_bf, n4);
    n4 = (long long)H4_ * I_ / 4;
    cast4_bf16_k<<<(unsigned)((n4 + 255) / 256), 256, 0, stream>>>(Wih1, Wih1_bf, n4);
    n4 = (long long)H4_ * H_ / 4;
    cast4_bf16_k<<<(unsigned)((n4 + 255) / 256), 256, 0, stream>>>(Whh1, Whh1_bf, n4);
    n4 = (long long)V_ * H_ / 4;
    cast4_bf16_k<<<(unsigned)((n4 + 255) / 256), 256, 0, stream>>>(Wout, Wout_bf, n4);
    int n = H4_ * 2 * H_;
    build_w2cat_k<<<(n + 255) / 256, 256, 0, stream>>>(Wih2, Whh2, W2c_bf);
    vec_add_k<<<(H4_ + 255) / 256, 256, 0, stream>>>(bih1, bhh1, b1s, H4_);
    vec_add_k<<<(H4_ + 255) / 256, 256, 0, stream>>>(bih2, bhh2, b2s, H4_);
    int nz = B_ * H_;
    zero_f32_k<<<(nz + 255) / 256, 256, 0, stream>>>(c1, nz);
    zero_f32_k<<<(nz + 255) / 256, 256, 0, stream>>>(c2, nz);
    zero_f32_k<<<(nz + 255) / 256, 256, 0, stream>>>(y2, nz);
    zero_bf16_k<<<(nz + 255) / 256, 256, 0, stream>>>(h1b, nz);
    int nz2 = B_ * 2 * H_;
    zero_bf16_k<<<(nz2 + 255) / 256, 256, 0, stream>>>(xcat, nz2);
  }

  const int BH = B_ * H_;
  const int cellBlocks = (BH + 255) / 256;

  // ---- Hoisted input projection: G1[b*T+t] = x[b,t,:] @ Wih1^T + b1 ----
  launch_gemm_big(stream, x_bf, I_, Wih1_bf, I_, b1s,
                  G1, H4_, B_ * T_, H4_, I_);

  // ---- Encoder: 80 steps of (lstm1, lstm2) ----
  for (int t = 0; t < T_; ++t) {
    // gates1 = h1 @ Whh1^T + G1[:,t,:]
    launch_gemm(stream, h1b, H_, Whh1_bf, H_,
                G1 + (long long)t * H4_, (long long)T_ * H4_, nullptr,
                gates1, H4_, B_, H4_, H_);
    // h1,c1 update; h1 -> h1b and xcat[:, :H] (lstm2 input = ys1[t])
    lstm_cell_k<<<cellBlocks, 256, 0, stream>>>(gates1, c1,
        h1b, H_, xcat, 2 * H_, nullptr, nullptr);
    // gates2 = [ys1[t] | h2] @ [Wih2|Whh2]^T + b2   (K = 2H)
    launch_gemm(stream, xcat, 2 * H_, W2c_bf, 2 * H_, nullptr, 0, b2s,
                gates2, H4_, B_, H4_, 2 * H_);
    // h2,c2 update; h2 -> xcat[:, H:]
    lstm_cell_k<<<cellBlocks, 256, 0, stream>>>(gates2, c2,
        xcat + H_, 2 * H_, nullptr, 0, nullptr, nullptr);
  }

  // ---- Extra lstm1 step with zero input: gates = h1 @ Whh1^T + b1 ----
  launch_gemm(stream, h1b, H_, Whh1_bf, H_, nullptr, 0, b1s,
              gates1, H4_, B_, H4_, H_);
  lstm_cell_k<<<cellBlocks, 256, 0, stream>>>(gates1, c1,
      h1b, H_, nullptr, 0, nullptr, nullptr);

  // ---- Decoder: 40 steps ----
  for (int s = 0; s < S_; ++s) {
    // lstm1 with zero input (skip the 4096-wide matmul entirely)
    launch_gemm(stream, h1b, H_, Whh1_bf, H_, nullptr, 0, b1s,
                gates1, H4_, B_, H4_, H_);
    // h1,c1 update; xcat[:, :H] = bf16(h1 + y2)
    lstm_cell_k<<<cellBlocks, 256, 0, stream>>>(gates1, c1,
        h1b, H_, xcat, 2 * H_, y2, nullptr);
    // gates2 = [h1+y2 | h2] @ [Wih2|Whh2]^T + b2
    launch_gemm(stream, xcat, 2 * H_, W2c_bf, 2 * H_, nullptr, 0, b2s,
                gates2, H4_, B_, H4_, 2 * H_);
    // h2,c2 update; h2 -> xcat[:, H:], y2 (fp32), feats row (b*S+s)
    lstm_cell_k<<<cellBlocks, 256, 0, stream>>>(gates2, c2,
        xcat + H_, 2 * H_, feats + (long long)s * H_, (long long)S_ * H_, nullptr, y2);
  }

  // ---- Vocab projection: out = feats @ Wout^T + bout  (5120 x 6000, K=256) ----
  launch_gemm_big(stream, feats, H_, Wout_bf, H_, bout,
                  out, V_, B_ * S_, V_, H_);
}
